// PyReg_77412490543830
// MI455X (gfx1250) — matrix operand order repaired and checked by reading, not compiled, not run on hardware
//
#include <hip/hip_runtime.h>

#define LRELU_SLOPE 0.2f
#define EPS_IN 1e-5f

typedef __attribute__((ext_vector_type(16))) _Float16 v16h;
typedef __attribute__((ext_vector_type(8)))  float    v8f;
typedef __attribute__((ext_vector_type(4)))  unsigned v4u;
typedef __attribute__((ext_vector_type(8)))  int      v8i_t;
typedef __attribute__((ext_vector_type(4)))  int      v4i_t;

// ======================= device kernels =======================

// Pack OIHW f32 weights into f16 implicit-GEMM layout Wp[Cout][K],
// K = ksz*ksz*Cin_pad, k = tap*Cin_pad + ci  (zero-padded channels).
__global__ void pack_weights_kernel(const float* __restrict__ w, _Float16* __restrict__ wp,
                                    int Cout, int Cin, int Cin_pad, int ksz)
{
    long long idx = (long long)blockIdx.x * blockDim.x + threadIdx.x;
    int K = ksz * ksz * Cin_pad;
    long long total = (long long)Cout * K;
    if (idx >= total) return;
    int co  = (int)(idx / K);
    int k   = (int)(idx - (long long)co * K);
    int tap = k / Cin_pad;
    int ci  = k - tap * Cin_pad;
    int ky  = tap / ksz, kx = tap - ky * ksz;
    float v = (ci < Cin) ? w[(((size_t)co * Cin + ci) * ksz + ky) * ksz + kx] : 0.0f;
    wp[idx] = (_Float16)v;
}

// Issue one TDM 2D tile load: rows x 32 f16 elements, row stride Kelems, into LDS.
// D# bitfields per CDNA5 ISA ch.8 (group0: count/lds/global/type, group1: dims/strides).
__device__ __forceinline__ void tdm_load_weights(unsigned lds_off,
                                                 const _Float16* gsrc,
                                                 int rows, int Kelems)
{
    unsigned long long ga = (unsigned long long)gsrc;
    v4u g0;
    g0[0] = 1u;                                        // count=1, user descriptor
    g0[1] = lds_off;                                   // lds_addr (bytes)
    g0[2] = (unsigned)(ga & 0xFFFFFFFFu);              // global_addr[31:0]
    g0[3] = (unsigned)((ga >> 32) & 0x1FFFFFFu) | (2u << 30);  // [56:32] | type=2
    v8i_t g1;
    g1[0] = (int)(1u << 16);                           // data_size=1 (2 bytes/elem)
    g1[1] = (int)(32u << 16);                          // tensor_dim0[15:0]=32 @bits63:48
    g1[2] = (int)(((unsigned)rows & 0xFFFFu) << 16);   // tensor_dim1[15:0] @bits95:80
    g1[3] = (int)(32u << 16);                          // tile_dim0=32 @bits127:112
    g1[4] = (int)((unsigned)rows & 0xFFFFu);           // tile_dim1=rows, tile_dim2=0
    g1[5] = (int)(unsigned)Kelems;                     // tensor_dim0_stride[31:0]
    g1[6] = 0;                                         // stride[47:32]=0, dim1_stride lo=0
    g1[7] = 0;
    v4i_t z4 = {};
#if defined(__clang_major__) && (__clang_major__ >= 23)
    v8i_t z8 = {};
    __builtin_amdgcn_tensor_load_to_lds(g0, g1, z4, z4, z8, 0);
#else
    __builtin_amdgcn_tensor_load_to_lds(g0, g1, z4, z4, 0);
#endif
}

// Implicit-GEMM conv, stride 1, optional lhs-dilation (transposed conv).
// 256 threads = 8 wave32; each wave owns NT 16(cout)x16(pixel) D tiles sharing one
// B fragment. Weights staged by the Tensor Data Mover, double-buffered in LDS.
template<int NT>
__global__ __launch_bounds__(256) void conv_wmma_kernel(
    const float* __restrict__ in, const _Float16* __restrict__ wp,
    const float* __restrict__ bias, float* __restrict__ out,
    int Cin, int Cin_pad, int Hin, int Win,
    int Cout, int Hout, int Wout, int ksz, int pad, int dil)
{
    __shared__ _Float16 lds_w[2][NT * 16 * 32];

    const int lane = threadIdx.x & 31;
    const int wave = threadIdx.x >> 5;
    const int n    = lane & 15;      // pixel column within tile / M row for A
    const int hi   = lane >> 4;      // half-wave selector
    const int HWo  = Hout * Wout;
    const int p    = (blockIdx.x * 8 + wave) * 16 + n;   // HWo is a multiple of 128
    const int co0  = blockIdx.y * (16 * NT);
    const int nb   = blockIdx.z;
    const int oy   = p / Wout, ox = p - oy * Wout;
    const int K    = ksz * ksz * Cin_pad;
    const size_t HWin = (size_t)Hin * Win;
    const float* inb = in + (size_t)nb * Cin * HWin;
    const int chunks = Cin_pad >> 5;
    const int S = ksz * ksz * chunks;
    const _Float16* wbase = wp + (size_t)co0 * K;

    v8f zero = {};
    v8f acc[NT];
    #pragma unroll
    for (int t = 0; t < NT; ++t) acc[t] = zero;

    // prologue: TDM-stage slab 0
    if (threadIdx.x == 0)
        tdm_load_weights((unsigned)(unsigned long long)&lds_w[0][0], wbase, NT * 16, K);

    int tap = 0, cchunk = 0;
    bool valid = false;
    const float* inpix = inb;
    for (int s = 0; s < S; ++s) {
        if (threadIdx.x == 0) {
            if (s + 1 < S) {   // double buffer: issue next slab, wait for current
                tdm_load_weights((unsigned)(unsigned long long)&lds_w[(s + 1) & 1][0],
                                 wbase + (size_t)(s + 1) * 32, NT * 16, K);
                __builtin_amdgcn_s_wait_tensorcnt(1);
            } else {
                __builtin_amdgcn_s_wait_tensorcnt(0);
            }
        }
        if (cchunk == 0) {   // new kernel tap: recompute input coordinate
            const int ky = tap / ksz, kx = tap - ky * ksz;
            int ty = oy + ky - pad, tx = ox + kx - pad;
            int iy, ix; bool vy, vx;
            if (dil == 1) {
                iy = ty; ix = tx;
                vy = (iy >= 0) && (iy < Hin);
                vx = (ix >= 0) && (ix < Win);
            } else {
                vy = (ty >= 0) && ((ty & 1) == 0); iy = ty >> 1; vy = vy && (iy < Hin);
                vx = (tx >= 0) && ((tx & 1) == 0); ix = tx >> 1; vx = vx && (ix < Win);
            }
            valid = vy && vx;
            inpix = inb + (valid ? ((size_t)iy * Win + ix) : 0);
            if (valid) __builtin_prefetch(inpix, 0, 1);   // global_prefetch_b8
        }
        __syncthreads();
        // ---- B fragment (32x16 f16): lanes 0-15 K rows 0..15, lanes 16-31 rows 16..31 ----
        const int ci0 = (cchunk << 5) + hi * 16;
        v16h b;
        {
            const float* q = inpix + (size_t)ci0 * HWin;
            #pragma unroll
            for (int j = 0; j < 16; ++j) {
                float v = (valid && (ci0 + j) < Cin) ? *q : 0.0f;
                b[j] = (_Float16)v;
                q += HWin;
            }
        }
        // ---- NT A fragments from LDS slab, NT WMMAs sharing the B fragment ----
        const _Float16* wbuf = lds_w[s & 1];
        #pragma unroll
        for (int t = 0; t < NT; ++t) {
            v16h a;
            const _Float16* lw = wbuf + (t * 16 + n) * 32;
            #pragma unroll
            for (int j = 0; j < 8; ++j) a[j]     = lw[hi * 16 + j];
            #pragma unroll
            for (int j = 0; j < 8; ++j) a[8 + j] = lw[hi * 16 + 8 + j];
            acc[t] = __builtin_amdgcn_wmma_f32_16x16x32_f16(
                         false, a, false, b, (short)0, acc[t], false, false);
        }
        __syncthreads();
        if (++cchunk == chunks) { cchunk = 0; ++tap; }
    }

    // D layout: col N = lane&15, row M = hi*8 + vgpr (per 16-subtile)
    float* outb = out + (size_t)nb * Cout * HWo;
    #pragma unroll
    for (int t = 0; t < NT; ++t) {
        #pragma unroll
        for (int r = 0; r < 8; ++r) {
            int m = t * 16 + hi * 8 + r;
            outb[(size_t)(co0 + m) * HWo + p] = acc[t][r] + bias[co0 + m];
        }
    }
}

// InstanceNorm (affine=False) + LeakyReLU, one block per (b,c) plane. In-place capable.
__global__ __launch_bounds__(256) void inorm_lrelu_kernel(const float* __restrict__ x,
                                                          float* __restrict__ y, int HW)
{
    const float* xp = x + (size_t)blockIdx.x * HW;
    float* yp = y + (size_t)blockIdx.x * HW;
    float s = 0.0f, s2 = 0.0f;
    for (int i = threadIdx.x; i < HW; i += 256) { float v = xp[i]; s += v; s2 = fmaf(v, v, s2); }
    __shared__ float sh0[256], sh1[256];
    sh0[threadIdx.x] = s; sh1[threadIdx.x] = s2;
    __syncthreads();
    for (int o = 128; o > 0; o >>= 1) {
        if (threadIdx.x < o) {
            sh0[threadIdx.x] += sh0[threadIdx.x + o];
            sh1[threadIdx.x] += sh1[threadIdx.x + o];
        }
        __syncthreads();
    }
    float inv  = 1.0f / (float)HW;
    float mean = sh0[0] * inv;
    float var  = sh1[0] * inv - mean * mean;
    float rs   = rsqrtf(var + EPS_IN);
    for (int i = threadIdx.x; i < HW; i += 256) {
        float v = (xp[i] - mean) * rs;
        yp[i] = (v > 0.0f) ? v : LRELU_SLOPE * v;
    }
}

// Direct 3x3 conv with Cout=2 (flow-generating "def" convs).
__global__ void defconv_kernel(const float* __restrict__ in, const float* __restrict__ w,
                               const float* __restrict__ bias, float* __restrict__ out,
                               int Cin, int H, int W, long long n)
{
    long long idx = (long long)blockIdx.x * blockDim.x + threadIdx.x;
    if (idx >= n) return;
    const size_t HW = (size_t)H * W;
    int hw = (int)(idx % (long long)HW);
    int bc = (int)(idx / (long long)HW);
    int co = bc & 1, b = bc >> 1;
    int y = hw / W, x = hw - y * W;
    const float* ib = in + (size_t)b * Cin * HW;
    float acc = bias[co];
    for (int ci = 0; ci < Cin; ++ci) {
        const float* ic = ib + (size_t)ci * HW;
        const float* wc = w + ((size_t)co * Cin + ci) * 9;
        #pragma unroll
        for (int ky = 0; ky < 3; ++ky) {
            int iy = y + ky - 1;
            if (iy < 0 || iy >= H) continue;
            #pragma unroll
            for (int kx = 0; kx < 3; ++kx) {
                int ix = x + kx - 1;
                if (ix < 0 || ix >= W) continue;
                acc = fmaf(wc[ky * 3 + kx], ic[(size_t)iy * W + ix], acc);
            }
        }
    }
    out[idx] = acc;
}

__device__ __forceinline__ float bsample(const float* __restrict__ s, int H, int W,
                                         float py, float px)
{
    float y0f = floorf(py), x0f = floorf(px);
    float wy = py - y0f, wx = px - x0f;
    int y0 = (int)y0f, x0 = (int)x0f;
    float acc = 0.0f;
    #pragma unroll
    for (int dy = 0; dy < 2; ++dy) {
        #pragma unroll
        for (int dx = 0; dx < 2; ++dx) {
            int yi = y0 + dy, xi = x0 + dx;
            if (yi >= 0 && yi < H && xi >= 0 && xi < W) {
                float ww = (dy ? wy : 1.0f - wy) * (dx ? wx : 1.0f - wx);
                acc = fmaf(s[(size_t)yi * W + xi], ww, acc);
            }
        }
    }
    return acc;
}

// out[b,c] = bilinear_sample(src[b,c], grid + flow[b])   (zero padding)
__global__ void warp_kernel(const float* __restrict__ src, const float* __restrict__ flow,
                            float* __restrict__ out, int C, int H, int W, long long n)
{
    long long idx = (long long)blockIdx.x * blockDim.x + threadIdx.x;
    if (idx >= n) return;
    const size_t HW = (size_t)H * W;
    int hw = (int)(idx % (long long)HW);
    int bc = (int)(idx / (long long)HW);
    int b = bc / C;
    int y = hw / W, x = hw - y * W;
    const float* fb = flow + (size_t)b * 2 * HW;
    float py = (float)y + fb[hw];
    float px = (float)x + fb[HW + hw];
    out[idx] = bsample(src + (size_t)bc * HW, H, W, py, px);
}

// out = warp(src2, f) + f   (src2 and f are 2-channel flow fields)
__global__ void warpadd_kernel(const float* __restrict__ src2, const float* __restrict__ f,
                               float* __restrict__ out, int H, int W, long long n)
{
    long long idx = (long long)blockIdx.x * blockDim.x + threadIdx.x;
    if (idx >= n) return;
    const size_t HW = (size_t)H * W;
    int hw = (int)(idx % (long long)HW);
    int bc = (int)(idx / (long long)HW);
    int b = bc >> 1;
    int y = hw / W, x = hw - y * W;
    const float* fb = f + (size_t)b * 2 * HW;
    float py = (float)y + fb[hw];
    float px = (float)x + fb[HW + hw];
    out[idx] = bsample(src2 + (size_t)bc * HW, H, W, py, px) + f[idx];
}

__global__ void scale_kernel(const float* __restrict__ x, float* __restrict__ y,
                             float s, long long n)
{
    long long idx = (long long)blockIdx.x * blockDim.x + threadIdx.x;
    if (idx < n) y[idx] = x[idx] * s;
}

// 2x bilinear upsample, align_corners=True, with pre-scale factor s.
__global__ void upsample2x_kernel(const float* __restrict__ in, float* __restrict__ out,
                                  int H, int W, float s, long long n)
{
    long long idx = (long long)blockIdx.x * blockDim.x + threadIdx.x;
    if (idx >= n) return;
    int Ho = 2 * H, Wo = 2 * W;
    size_t HWo = (size_t)Ho * Wo;
    int hw = (int)(idx % (long long)HWo);
    size_t pl = (size_t)(idx / (long long)HWo);
    int oy = hw / Wo, ox = hw - oy * Wo;
    float ys = (float)oy * (float)(H - 1) / (float)(Ho - 1);
    float xs = (float)ox * (float)(W - 1) / (float)(Wo - 1);
    int y0 = (int)floorf(ys); y0 = y0 < 0 ? 0 : (y0 > H - 2 ? H - 2 : y0);
    int x0 = (int)floorf(xs); x0 = x0 < 0 ? 0 : (x0 > W - 2 ? W - 2 : x0);
    float wy = ys - (float)y0, wx = xs - (float)x0;
    const float* ip = in + pl * (size_t)H * W;
    float r0 = ip[(size_t)y0 * W + x0] * (1.0f - wx) + ip[(size_t)y0 * W + x0 + 1] * wx;
    float r1 = ip[(size_t)(y0 + 1) * W + x0] * (1.0f - wx) + ip[(size_t)(y0 + 1) * W + x0 + 1] * wx;
    out[idx] = (r0 * (1.0f - wy) + r1 * wy) * s;
}

// copy src[B,C,HW] into dst[B,Ctot,HW] at channel offset coff
__global__ void concat_copy_kernel(const float* __restrict__ src, float* __restrict__ dst,
                                   int C, long long HW, int Ctot, int coff, long long n)
{
    long long idx = (long long)blockIdx.x * blockDim.x + threadIdx.x;
    if (idx >= n) return;
    long long chw = (long long)C * HW;
    int b = (int)(idx / chw);
    long long r = idx - (long long)b * chw;
    int c = (int)(r / HW);
    long long hw = r - (long long)c * HW;
    dst[((size_t)b * Ctot + coff + c) * (size_t)HW + hw] = src[idx];
}

// ======================= host orchestration =======================

namespace {

struct Arena { char* p; size_t off; };
static inline float* af(Arena& a, size_t nfloats) {
    float* r = (float*)(a.p + a.off);
    a.off += ((nfloats * sizeof(float)) + 255) & ~(size_t)255;
    return r;
}
static inline _Float16* ah(Arena& a, size_t nhalf) {
    _Float16* r = (_Float16*)(a.p + a.off);
    a.off += ((nhalf * sizeof(_Float16)) + 255) & ~(size_t)255;
    return r;
}

struct PackedConv { const _Float16* wp; const float* bias; int Cout, Cin, Cin_pad, ksz; };

static PackedConv pack_conv(hipStream_t st, Arena& ar, const float* w, const float* b,
                            int Cout, int Cin, int ksz)
{
    int Cin_pad = (Cin + 31) & ~31;
    size_t n = (size_t)Cout * ksz * ksz * Cin_pad;
    _Float16* wp = ah(ar, n);
    unsigned blocks = (unsigned)((n + 255) / 256);
    pack_weights_kernel<<<blocks, 256, 0, st>>>(w, wp, Cout, Cin, Cin_pad, ksz);
    return PackedConv{wp, b, Cout, Cin, Cin_pad, ksz};
}

static void run_conv(hipStream_t st, const PackedConv& pc, const float* in, float* out,
                     int B, int Hin, int Win, int Hout, int Wout, int pad, int dil)
{
    int nt = (pc.Cout % 64 == 0) ? 4 : (pc.Cout % 48 == 0) ? 3
           : (pc.Cout % 32 == 0) ? 2 : 1;
    dim3 grid((unsigned)(Hout * Wout / 128), (unsigned)(pc.Cout / (16 * nt)), (unsigned)B);
    switch (nt) {
    case 4:
        conv_wmma_kernel<4><<<grid, 256, 0, st>>>(in, pc.wp, pc.bias, out,
            pc.Cin, pc.Cin_pad, Hin, Win, pc.Cout, Hout, Wout, pc.ksz, pad, dil);
        break;
    case 3:
        conv_wmma_kernel<3><<<grid, 256, 0, st>>>(in, pc.wp, pc.bias, out,
            pc.Cin, pc.Cin_pad, Hin, Win, pc.Cout, Hout, Wout, pc.ksz, pad, dil);
        break;
    case 2:
        conv_wmma_kernel<2><<<grid, 256, 0, st>>>(in, pc.wp, pc.bias, out,
            pc.Cin, pc.Cin_pad, Hin, Win, pc.Cout, Hout, Wout, pc.ksz, pad, dil);
        break;
    default:
        conv_wmma_kernel<1><<<grid, 256, 0, st>>>(in, pc.wp, pc.bias, out,
            pc.Cin, pc.Cin_pad, Hin, Win, pc.Cout, Hout, Wout, pc.ksz, pad, dil);
        break;
    }
}

static void run_cib(hipStream_t st, const PackedConv& pc, const float* in, float* out,
                    int B, int H)
{
    run_conv(st, pc, in, out, B, H, H, H, H, 1, 1);
    inorm_lrelu_kernel<<<(unsigned)(B * pc.Cout), 256, 0, st>>>(out, out, H * H);
}

static void run_upblock(hipStream_t st, const PackedConv& pc, const float* in, float* out,
                        int B, int Hin)
{
    run_conv(st, pc, in, out, B, Hin, Hin, 2 * Hin, 2 * Hin, 2, 2);
    inorm_lrelu_kernel<<<(unsigned)(B * pc.Cout), 256, 0, st>>>(out, out, 4 * Hin * Hin);
}

static inline unsigned ew_blocks(long long n) { return (unsigned)((n + 255) / 256); }

static void run_defconv(hipStream_t st, const float* w, const float* b, const float* in,
                        float* out, int B, int Cin, int H)
{
    long long n = (long long)B * 2 * H * H;
    defconv_kernel<<<ew_blocks(n), 256, 0, st>>>(in, w, b, out, Cin, H, H, n);
}

static void run_warp(hipStream_t st, const float* src, const float* flow, float* out,
                     int B, int C, int H)
{
    long long n = (long long)B * C * H * H;
    warp_kernel<<<ew_blocks(n), 256, 0, st>>>(src, flow, out, C, H, H, n);
}

static void run_warpadd(hipStream_t st, const float* src2, const float* f, float* out,
                        int B, int H)
{
    long long n = (long long)B * 2 * H * H;
    warpadd_kernel<<<ew_blocks(n), 256, 0, st>>>(src2, f, out, H, H, n);
}

// VecInt scaling-and-squaring: v /= 128; 7x: v = v + warp(v,v). Returns final buffer.
static float* run_vecint(hipStream_t st, float* v, float* tmp, int B, int H)
{
    long long n = (long long)B * 2 * H * H;
    scale_kernel<<<ew_blocks(n), 256, 0, st>>>(v, v, 1.0f / 128.0f, n);
    float* cur = v; float* oth = tmp;
    for (int i = 0; i < 7; ++i) {
        warpadd_kernel<<<ew_blocks(n), 256, 0, st>>>(cur, cur, oth, H, H, n);
        float* t = cur; cur = oth; oth = t;
    }
    return cur;
}

static void run_upsample(hipStream_t st, const float* in, float* out, int planes, int H,
                         float s)
{
    long long n = (long long)planes * 4 * H * H;
    upsample2x_kernel<<<ew_blocks(n), 256, 0, st>>>(in, out, H, H, s, n);
}

static void run_concat(hipStream_t st, const float* src, float* dst, int B, int C,
                       long long HW, int Ctot, int coff)
{
    long long n = (long long)B * C * HW;
    concat_copy_kernel<<<ew_blocks(n), 256, 0, st>>>(src, dst, C, HW, Ctot, coff, n);
}

} // namespace

extern "C" void kernel_launch(void* const* d_in, const int* in_sizes, int n_in,
                              void* d_out, int out_size, void* d_ws, size_t ws_size,
                              hipStream_t stream)
{
    (void)in_sizes; (void)n_in; (void)out_size; (void)ws_size;
    const int B = 2;
    const int H1 = 256, H2 = 128, H3 = 64, H4 = 32;
    const size_t HW1 = 65536, HW2 = 16384, HW3 = 4096, HW4 = 1024;

    const float* M1 = (const float*)d_in[0];
    const float* F1 = (const float*)d_in[1];
    const float* M2 = (const float*)d_in[2];
    const float* F2 = (const float*)d_in[3];
    const float* M3 = (const float*)d_in[4];
    const float* F3 = (const float*)d_in[5];
    const float* M4 = (const float*)d_in[6];
    const float* F4 = (const float*)d_in[7];
    // d_in[8] = moving (unused by the reference network)
#define PAR(i) ((const float*)d_in[(i)])

    Arena ar; ar.p = (char*)d_ws; ar.off = 0;

    // -------- pack all WMMA conv weights to f16 GEMM layout --------
    PackedConv Pconv0 = pack_conv(stream, ar, PAR(9),  PAR(10), 16, 64, 3);
    PackedConv Pconv1 = pack_conv(stream, ar, PAR(11), PAR(12), 32, 128, 3);
    PackedConv Pconv2 = pack_conv(stream, ar, PAR(13), PAR(14), 64, 256, 3);
    PackedConv Pconv3 = pack_conv(stream, ar, PAR(15), PAR(16), 128, 512, 3);
    PackedConv Pcc4a  = pack_conv(stream, ar, PAR(17), PAR(18), 128, 256, 3);
    PackedConv Pcc4b  = pack_conv(stream, ar, PAR(19), PAR(20), 128, 128, 3);
    PackedConv Pdc4a  = pack_conv(stream, ar, PAR(21), PAR(22), 128, 384, 3);
    PackedConv Pdc4b  = pack_conv(stream, ar, PAR(23), PAR(24), 128, 128, 3);
    PackedConv Pcc3a  = pack_conv(stream, ar, PAR(25), PAR(26), 192, 192, 3);
    PackedConv Pcc3b  = pack_conv(stream, ar, PAR(27), PAR(28), 192, 192, 3);
    PackedConv Pdc3   = pack_conv(stream, ar, PAR(29), PAR(30), 64, 192, 3);
    PackedConv Pcc2a  = pack_conv(stream, ar, PAR(31), PAR(32), 96, 96, 3);
    PackedConv Pcc2b  = pack_conv(stream, ar, PAR(33), PAR(34), 96, 96, 3);
    PackedConv Pdc2   = pack_conv(stream, ar, PAR(35), PAR(36), 32, 96, 3);
    PackedConv Pcc1a  = pack_conv(stream, ar, PAR(37), PAR(38), 48, 48, 3);
    PackedConv Pcc1b  = pack_conv(stream, ar, PAR(39), PAR(40), 48, 48, 3);
    PackedConv Pup3   = pack_conv(stream, ar, PAR(41), PAR(42), 64, 128, 4);
    PackedConv Pup2   = pack_conv(stream, ar, PAR(43), PAR(44), 32, 192, 4);
    PackedConv Pup1   = pack_conv(stream, ar, PAR(45), PAR(46), 16, 96, 4);
    const float *def4w = PAR(47), *def4b = PAR(48);
    const float *def3w = PAR(49), *def3b = PAR(50);
    const float *def2w = PAR(51), *def2b = PAR(52);
    const float *def1w = PAR(53), *def1b = PAR(54);

    // -------- encoders (ConvInsBlock on each pyramid level) --------
    float* eM1 = af(ar, (size_t)B * 16 * HW1);  run_cib(stream, Pconv0, M1, eM1, B, H1);
    float* eF1 = af(ar, (size_t)B * 16 * HW1);  run_cib(stream, Pconv0, F1, eF1, B, H1);
    float* eM2 = af(ar, (size_t)B * 32 * HW2);  run_cib(stream, Pconv1, M2, eM2, B, H2);
    float* eF2 = af(ar, (size_t)B * 32 * HW2);  run_cib(stream, Pconv1, F2, eF2, B, H2);
    float* eM3 = af(ar, (size_t)B * 64 * HW3);  run_cib(stream, Pconv2, M3, eM3, B, H3);
    float* eF3 = af(ar, (size_t)B * 64 * HW3);  run_cib(stream, Pconv2, F3, eF3, B, H3);
    float* eM4 = af(ar, (size_t)B * 128 * HW4); run_cib(stream, Pconv3, M4, eM4, B, H4);
    float* eF4 = af(ar, (size_t)B * 128 * HW4); run_cib(stream, Pconv3, F4, eF4, B, H4);

    // ================= level 4 (1/8 res) =================
    float* D3     = af(ar, (size_t)B * 64 * HW3);   // upblock output for level 3
    float* flowL3 = af(ar, (size_t)B * 2 * HW3);    // upsampled flow for level 3
    size_t mark4 = ar.off;
    {
        float* cat0 = af(ar, (size_t)B * 256 * HW4);
        run_concat(stream, eF4, cat0, B, 128, HW4, 256, 0);
        run_concat(stream, eM4, cat0, B, 128, HW4, 256, 128);
        float* t0 = af(ar, (size_t)B * 128 * HW4);  run_cib(stream, Pcc4a, cat0, t0, B, H4);
        float* C4 = af(ar, (size_t)B * 128 * HW4);  run_cib(stream, Pcc4b, t0, C4, B, H4);
        float* fl  = af(ar, (size_t)B * 2 * HW4);
        float* flb = af(ar, (size_t)B * 2 * HW4);
        run_defconv(stream, def4w, def4b, C4, fl, B, 128, H4);
        float* flow = run_vecint(stream, fl, flb, B, H4);
        float* warped = af(ar, (size_t)B * 128 * HW4);
        run_warp(stream, eM4, flow, warped, B, 128, H4);
        float* cat1 = af(ar, (size_t)B * 384 * HW4);
        run_concat(stream, eF4,    cat1, B, 128, HW4, 384, 0);
        run_concat(stream, warped, cat1, B, 128, HW4, 384, 128);
        run_concat(stream, C4,     cat1, B, 128, HW4, 384, 256);
        float* t1  = af(ar, (size_t)B * 128 * HW4); run_cib(stream, Pdc4a, cat1, t1, B, H4);
        float* C4b = af(ar, (size_t)B * 128 * HW4); run_cib(stream, Pdc4b, t1, C4b, B, H4);
        float* wf  = af(ar, (size_t)B * 2 * HW4);
        float* wfb = af(ar, (size_t)B * 2 * HW4);
        run_defconv(stream, def4w, def4b, C4b, wf, B, 128, H4);
        float* w = run_vecint(stream, wf, wfb, B, H4);
        run_upblock(stream, Pup3, C4b, D3, B, H4);
        float* comp = af(ar, (size_t)B * 2 * HW4);
        run_warpadd(stream, flow, w, comp, B, H4);
        run_upsample(stream, comp, flowL3, B * 2, H4, 2.0f);
    }
    ar.off = mark4;

    // ================= level 3 =================
    float* D2     = af(ar, (size_t)B * 32 * HW2);
    float* flowL2 = af(ar, (size_t)B * 2 * HW2);
    size_t mark3 = ar.off;
    {
        float* warped = af(ar, (size_t)B * 64 * HW3);
        run_warp(stream, eM3, flowL3, warped, B, 64, H3);
        float* cat0 = af(ar, (size_t)B * 192 * HW3);
        run_concat(stream, eF3,    cat0, B, 64, HW3, 192, 0);
        run_concat(stream, warped, cat0, B, 64, HW3, 192, 64);
        run_concat(stream, D3,     cat0, B, 64, HW3, 192, 128);
        float* t0 = af(ar, (size_t)B * 192 * HW3);  run_cib(stream, Pcc3a, cat0, t0, B, H3);
        float* C3 = af(ar, (size_t)B * 192 * HW3);  run_cib(stream, Pcc3b, t0, C3, B, H3);
        float* wf  = af(ar, (size_t)B * 2 * HW3);
        float* wfb = af(ar, (size_t)B * 2 * HW3);
        run_defconv(stream, def3w, def3b, C3, wf, B, 192, H3);
        float* w = run_vecint(stream, wf, wfb, B, H3);
        float* flowb = af(ar, (size_t)B * 2 * HW3);
        run_warpadd(stream, flowL3, w, flowb, B, H3);
        run_warp(stream, eM3, flowb, warped, B, 64, H3);
        float* D3b = af(ar, (size_t)B * 64 * HW3);  run_cib(stream, Pdc3, C3, D3b, B, H3);
        float* cat1 = af(ar, (size_t)B * 192 * HW3);
        run_concat(stream, eF3,    cat1, B, 64, HW3, 192, 0);
        run_concat(stream, warped, cat1, B, 64, HW3, 192, 64);
        run_concat(stream, D3b,    cat1, B, 64, HW3, 192, 128);
        float* t1  = af(ar, (size_t)B * 192 * HW3); run_cib(stream, Pcc3a, cat1, t1, B, H3);
        float* C3b = af(ar, (size_t)B * 192 * HW3); run_cib(stream, Pcc3b, t1, C3b, B, H3);
        float* wf2  = af(ar, (size_t)B * 2 * HW3);
        float* wfb2 = af(ar, (size_t)B * 2 * HW3);
        run_defconv(stream, def3w, def3b, C3b, wf2, B, 192, H3);
        float* w2 = run_vecint(stream, wf2, wfb2, B, H3);
        run_upblock(stream, Pup2, C3b, D2, B, H3);
        float* comp = af(ar, (size_t)B * 2 * HW3);
        run_warpadd(stream, flowb, w2, comp, B, H3);
        run_upsample(stream, comp, flowL2, B * 2, H3, 2.0f);
    }
    ar.off = mark3;

    // ================= level 2 =================
    float* D1     = af(ar, (size_t)B * 16 * HW1);
    float* flowL1 = af(ar, (size_t)B * 2 * HW1);
    size_t mark2 = ar.off;
    {
        float* warped = af(ar, (size_t)B * 32 * HW2);
        run_warp(stream, eM2, flowL2, warped, B, 32, H2);
        float* cat0 = af(ar, (size_t)B * 96 * HW2);
        run_concat(stream, eF2,    cat0, B, 32, HW2, 96, 0);
        run_concat(stream, warped, cat0, B, 32, HW2, 96, 32);
        run_concat(stream, D2,     cat0, B, 32, HW2, 96, 64);
        float* t0 = af(ar, (size_t)B * 96 * HW2);   run_cib(stream, Pcc2a, cat0, t0, B, H2);
        float* C2 = af(ar, (size_t)B * 96 * HW2);   run_cib(stream, Pcc2b, t0, C2, B, H2);
        float* wf  = af(ar, (size_t)B * 2 * HW2);
        float* wfb = af(ar, (size_t)B * 2 * HW2);
        run_defconv(stream, def2w, def2b, C2, wf, B, 96, H2);
        float* w = run_vecint(stream, wf, wfb, B, H2);
        float* flowb = af(ar, (size_t)B * 2 * HW2);
        run_warpadd(stream, flowL2, w, flowb, B, H2);
        run_warp(stream, eM2, flowb, warped, B, 32, H2);
        float* D2b = af(ar, (size_t)B * 32 * HW2);  run_cib(stream, Pdc2, C2, D2b, B, H2);
        float* cat1 = af(ar, (size_t)B * 96 * HW2);
        run_concat(stream, eF2,    cat1, B, 32, HW2, 96, 0);
        run_concat(stream, warped, cat1, B, 32, HW2, 96, 32);
        run_concat(stream, D2b,    cat1, B, 32, HW2, 96, 64);
        float* t1  = af(ar, (size_t)B * 96 * HW2);  run_cib(stream, Pcc2a, cat1, t1, B, H2);
        float* C2b = af(ar, (size_t)B * 96 * HW2);  run_cib(stream, Pcc2b, t1, C2b, B, H2);
        float* wf2  = af(ar, (size_t)B * 2 * HW2);
        float* wfb2 = af(ar, (size_t)B * 2 * HW2);
        run_defconv(stream, def2w, def2b, C2b, wf2, B, 96, H2);
        float* w2 = run_vecint(stream, wf2, wfb2, B, H2);
        run_upblock(stream, Pup1, C2b, D1, B, H2);
        float* comp = af(ar, (size_t)B * 2 * HW2);
        run_warpadd(stream, flowb, w2, comp, B, H2);
        run_upsample(stream, comp, flowL1, B * 2, H2, 2.0f);
    }
    ar.off = mark2;

    // ================= level 1 (full res) =================
    {
        float* warped = af(ar, (size_t)B * 16 * HW1);
        run_warp(stream, eM1, flowL1, warped, B, 16, H1);
        float* cat0 = af(ar, (size_t)B * 48 * HW1);
        run_concat(stream, eF1,    cat0, B, 16, HW1, 48, 0);
        run_concat(stream, warped, cat0, B, 16, HW1, 48, 16);
        run_concat(stream, D1,     cat0, B, 16, HW1, 48, 32);
        float* t0 = af(ar, (size_t)B * 48 * HW1);   run_cib(stream, Pcc1a, cat0, t0, B, H1);
        float* C1 = af(ar, (size_t)B * 48 * HW1);   run_cib(stream, Pcc1b, t0, C1, B, H1);
        float* wf  = af(ar, (size_t)B * 2 * HW1);
        float* wfb = af(ar, (size_t)B * 2 * HW1);
        run_defconv(stream, def1w, def1b, C1, wf, B, 48, H1);
        float* w = run_vecint(stream, wf, wfb, B, H1);
        float* flowF = af(ar, (size_t)B * 2 * HW1);
        run_warpadd(stream, flowL1, w, flowF, B, H1);

        // outputs: [warped(B,16,256,256) | flow(B,2,256,256)]
        float* outp = (float*)d_out;
        run_warp(stream, eM1, flowF, outp, B, 16, H1);
        long long nf = (long long)B * 2 * HW1;
        scale_kernel<<<ew_blocks(nf), 256, 0, stream>>>(flowF, outp + (size_t)B * 16 * HW1,
                                                        1.0f, nf);
    }
#undef PAR
}
